// GATv2Message_10677288698115
// MI455X (gfx1250) — matrix operand order repaired and checked by reading, not compiled, not run on hardware
//
#include <hip/hip_runtime.h>

#define D_DIM 128
#define H_DIM 4
#define C_DIM 32
#define NEG_SLOPE 0.2f

typedef __attribute__((ext_vector_type(16))) __bf16 v16bf;
typedef __attribute__((ext_vector_type(2)))  __bf16 v2bf;
typedef __attribute__((ext_vector_type(8)))  float  v8f;

// ---------------------------------------------------------------------------
// One GEMM pass of the fused triple GEMM. Inlined with direct kernel-arg
// pointers so the backend keeps global (not flat) memory ops.
// Bswz: [kt][nt][lane][j] per-lane contiguous 16-bf16 B fragments (32 KB).
// ---------------------------------------------------------------------------
__device__ __forceinline__
void gemm_one(const v16bf (&aT)[4], __bf16 (*Bswz)[8][32][16],
              const float* __restrict__ W, const float* __restrict__ bvec,
              float* __restrict__ Y,
              int tid, int lane, int ln, int hi, int m0, int n)
{
    // Build swizzled B image: thread owns a (k,k+1) pair of one column ->
    // one packed v2bf -> one ds_store_b32. Coalesced global reads.
    for (int p = tid; p < 64 * 128; p += 256) {
        const int kp  = p >> 7;            // k = 2*kp
        const int nn  = p & 127;
        const int k   = kp << 1;
        const int kt  = k >> 5;
        const int kk  = k & 31;
        const int hi2 = (kk >> 3) & 1;
        const int j   = (kk & 7) | ((kk >> 4) << 3);   // even
        const int ln2 = (hi2 << 4) | (nn & 15);
        const int nt  = nn >> 4;
        v2bf pk;
        pk[0] = (__bf16)W[(size_t)k * D_DIM + nn];
        pk[1] = (__bf16)W[(size_t)(k + 1) * D_DIM + nn];
        *reinterpret_cast<v2bf*>(&Bswz[kt][nt][ln2][j]) = pk;
    }
    __syncthreads();

    // m0 is wave-uniform; pin it to an SGPR so the store guards become
    // scalar branches (s_cmp/s_cbranch) instead of exec-mask dances.
    const int  m0u    = __builtin_amdgcn_readfirstlane(m0);
    const bool active = (m0u < n);
    const bool full   = (m0u + 16 <= n);
    float* __restrict__ ybase = Y + (size_t)(m0u + hi * 8) * D_DIM + ln;

    #pragma unroll
    for (int nt = 0; nt < 8; ++nt) {
        v8f acc = {};
        #pragma unroll
        for (int kt = 0; kt < 4; ++kt) {
            const v16bf bT =
                *reinterpret_cast<const v16bf*>(&Bswz[kt][nt][lane][0]);
            acc = __builtin_amdgcn_wmma_f32_16x16x32_bf16(
                false, aT[kt], false, bT, (short)0, acc, false, false);
        }
        const float badd = bvec[nt * 16 + ln];
        if (full) {
            // all 16 rows valid: plain immediate-offset global stores
            #pragma unroll
            for (int v = 0; v < 8; ++v)
                ybase[v * D_DIM + nt * 16] = acc[v] + badd;
        } else if (active) {
            #pragma unroll
            for (int v = 0; v < 8; ++v) {
                const int row = m0u + hi * 8 + v;
                if (row < n)
                    ybase[v * D_DIM + nt * 16] = acc[v] + badd;
            }
        }
    }
    __syncthreads();   // all waves done with Bswz before next refill
}

// ---------------------------------------------------------------------------
// Fused triple GEMM: Yi[n,128] = X[n,128] @ Wi[128,128] + bi   for i = 0..2
// bf16 WMMA, f32 accumulate; A (X rows) loaded once, reused for all three Ws.
// ---------------------------------------------------------------------------
__global__ __launch_bounds__(256)
void gemm128x3_wmma_bf16(const float* __restrict__ X,
                         const float* __restrict__ W0, const float* __restrict__ b0, float* __restrict__ Y0,
                         const float* __restrict__ W1, const float* __restrict__ b1, float* __restrict__ Y1,
                         const float* __restrict__ W2, const float* __restrict__ b2, float* __restrict__ Y2,
                         int n)
{
    __shared__ __bf16 Bswz[4][8][32][16];

    const int tid  = threadIdx.x;
    const int wave = tid >> 5;
    const int lane = tid & 31;
    const int ln   = lane & 15;
    const int hi   = lane >> 4;
    const int m0   = blockIdx.x * 128 + wave * 16;     // 16-row strip per wave
    const int m    = m0 + ln;
    const float* xrow = X + (size_t)((m < n) ? m : 0) * D_DIM;

    // Hoist all 4 A-tiles (16x32 bf16, ISA 16-bit A layout) into registers.
    v16bf aT[4];
    {
        const int ksel = hi * 8;
        #pragma unroll
        for (int kt = 0; kt < 4; ++kt) {
            const int k0 = kt * 32;
            #pragma unroll
            for (int v = 0; v < 8; ++v) {
                const int k = k0 + ((v >> 2) << 4) + ksel + ((v & 3) << 1);
                aT[kt][2 * v]     = (__bf16)xrow[k];
                aT[kt][2 * v + 1] = (__bf16)xrow[k + 1];
            }
        }
    }

    gemm_one(aT, Bswz, W0, b0, Y0, tid, lane, ln, hi, m0, n);
    gemm_one(aT, Bswz, W1, b1, Y1, tid, lane, ln, hi, m0, n);
    gemm_one(aT, Bswz, W2, b2, Y2, tid, lane, ln, hi, m0, n);
}

// ---------------------------------------------------------------------------
// Monotonic f32 <-> ordered-u32 encoding for atomic segment max
// ---------------------------------------------------------------------------
__device__ __forceinline__ unsigned enc_f32(float f) {
    unsigned u = __float_as_uint(f);
    return (u & 0x80000000u) ? ~u : (u | 0x80000000u);
}
__device__ __forceinline__ float dec_f32(unsigned u) {
    unsigned b = (u & 0x80000000u) ? (u ^ 0x80000000u) : ~u;
    return __uint_as_float(b);
}
#define ENC_NEG_INF 0x007FFFFFu   // enc_f32(-inf)

__global__ __launch_bounds__(256)
void init_buffers(float* __restrict__ accum, unsigned* __restrict__ amax,
                  float* __restrict__ denom, size_t nd, size_t nh)
{
    const size_t i = (size_t)blockIdx.x * 256 + threadIdx.x;
    if (i < nd) accum[i] = 0.0f;
    if (i < nh) { amax[i] = ENC_NEG_INF; denom[i] = 0.0f; }
}

// ---------------------------------------------------------------------------
// Pass A: alpha[e,h] = att_h . leaky_relu(xl[src] + xr[dst]); segment max.
// One wave per edge; lane owns 4 channels; 8-lane head-group reduction.
// ---------------------------------------------------------------------------
__global__ __launch_bounds__(256)
void edge_score(const float* __restrict__ xl, const float* __restrict__ xr,
                const long long* __restrict__ src, const long long* __restrict__ dst,
                const float* __restrict__ att, float* __restrict__ alpha,
                unsigned* __restrict__ amax, int e_count)
{
    const int e = blockIdx.x * 8 + (threadIdx.x >> 5);
    if (e >= e_count) return;
    const int lane = threadIdx.x & 31;

    const long long s = src[e];
    const long long d = dst[e];
    const float4 xs = reinterpret_cast<const float4*>(xl + (size_t)s * D_DIM)[lane];
    const float4 xd = reinterpret_cast<const float4*>(xr + (size_t)d * D_DIM)[lane];
    float4 v;
    v.x = xs.x + xd.x; v.y = xs.y + xd.y; v.z = xs.z + xd.z; v.w = xs.w + xd.w;
    v.x = v.x > 0.f ? v.x : NEG_SLOPE * v.x;
    v.y = v.y > 0.f ? v.y : NEG_SLOPE * v.y;
    v.z = v.z > 0.f ? v.z : NEG_SLOPE * v.z;
    v.w = v.w > 0.f ? v.w : NEG_SLOPE * v.w;

    const float4 aw = reinterpret_cast<const float4*>(att)[lane]; // flat [H*C]
    float p = v.x * aw.x + v.y * aw.y + v.z * aw.z + v.w * aw.w;

    p += __shfl_xor(p, 4, 32);
    p += __shfl_xor(p, 2, 32);
    p += __shfl_xor(p, 1, 32);

    if ((lane & 7) == 0) {
        const int h = lane >> 3;
        alpha[(size_t)e * H_DIM + h] = p;
        atomicMax(&amax[(size_t)d * H_DIM + h], enc_f32(p));
    }
}

// ---------------------------------------------------------------------------
// Pass B (fused): ex = exp(alpha - max); denom[d,h] += ex;
//                 accum[d,:] += ex * xl[src,:]   (normalized in finalize)
// ---------------------------------------------------------------------------
__global__ __launch_bounds__(256)
void edge_scatter(const float* __restrict__ xl,
                  const long long* __restrict__ src, const long long* __restrict__ dst,
                  const float* __restrict__ alpha, const unsigned* __restrict__ amax,
                  float* __restrict__ denom, float* __restrict__ accum, int e_count)
{
    const int e = blockIdx.x * 8 + (threadIdx.x >> 5);
    if (e >= e_count) return;
    const int lane = threadIdx.x & 31;
    const int h = lane >> 3;

    const long long s = src[e];
    const long long d = dst[e];

    const float a  = alpha[(size_t)e * H_DIM + h];
    float mx = dec_f32(amax[(size_t)d * H_DIM + h]);
    if ((__float_as_uint(mx) & 0x7F800000u) == 0x7F800000u) mx = 0.0f; // !isfinite
    const float ex = __expf(a - mx);

    if ((lane & 7) == 0)
        atomicAdd(&denom[(size_t)d * H_DIM + h], ex);

    const float4 xs = reinterpret_cast<const float4*>(xl + (size_t)s * D_DIM)[lane];
    float* out = accum + (size_t)d * D_DIM + lane * 4;
    atomicAdd(out + 0, ex * xs.x);
    atomicAdd(out + 1, ex * xs.y);
    atomicAdd(out + 2, ex * xs.z);
    atomicAdd(out + 3, ex * xs.w);
}

// ---------------------------------------------------------------------------
// out = elu(accum / (denom + 1e-16) + xres)       (xres already has bias)
// ---------------------------------------------------------------------------
__global__ __launch_bounds__(256)
void finalize(const float* __restrict__ accum, const float* __restrict__ denom,
              const float* __restrict__ xres, float* __restrict__ out, size_t nd)
{
    const size_t i = (size_t)blockIdx.x * 256 + threadIdx.x;
    if (i >= nd) return;
    const size_t node = i >> 7;
    const int    h    = ((int)i & 127) >> 5;
    const float  ds   = denom[node * H_DIM + h] + 1e-16f;
    const float  v    = accum[i] / ds + xres[i];
    out[i] = v > 0.0f ? v : (__expf(v) - 1.0f);
}

// ---------------------------------------------------------------------------
extern "C" void kernel_launch(void* const* d_in, const int* in_sizes, int n_in,
                              void* d_out, int out_size, void* d_ws, size_t ws_size,
                              hipStream_t stream)
{
    const float*     x    = (const float*)d_in[0];
    const long long* ei   = (const long long*)d_in[1];   // int64 [2,E]
    const float*     Wl   = (const float*)d_in[2];
    const float*     bl   = (const float*)d_in[3];
    const float*     Wr   = (const float*)d_in[4];
    const float*     br   = (const float*)d_in[5];
    const float*     att  = (const float*)d_in[6];
    const float*     Wres = (const float*)d_in[7];
    const float*     bias = (const float*)d_in[8];

    const int N = in_sizes[0] / D_DIM;
    const int E = in_sizes[1] / 2;
    const long long* src = ei;
    const long long* dst = ei + E;

    // workspace carve-up (~114 MB)
    const size_t nd = (size_t)N * D_DIM;
    const size_t nh = (size_t)N * H_DIM;
    float* ws    = (float*)d_ws;
    float* xl    = ws;                 ws += nd;
    float* xr    = ws;                 ws += nd;
    float* xres  = ws;                 ws += nd;
    float* accum = ws;                 ws += nd;
    float* x1    = ws;                 ws += nd;
    float* alpha = ws;                 ws += (size_t)E * H_DIM;
    float* denom = ws;                 ws += nh;
    unsigned* amax = (unsigned*)ws;

    const dim3 gemmGrid((N + 127) / 128);
    const int  edgeBlocks = (E + 7) / 8;
    const int  ndBlocks   = (int)((nd + 255) / 256);

    const float* xin = x;
    for (int l = 0; l < 2; ++l) {
        const size_t wOff = (size_t)l * D_DIM * D_DIM;
        const size_t vOff = (size_t)l * D_DIM;

        gemm128x3_wmma_bf16<<<gemmGrid, 256, 0, stream>>>(
            xin,
            Wl + wOff,   bl + vOff,   xl,
            Wr + wOff,   br + vOff,   xr,
            Wres + wOff, bias + vOff, xres,
            N);

        init_buffers<<<ndBlocks, 256, 0, stream>>>(accum, amax, denom, nd, nh);

        edge_score  <<<edgeBlocks, 256, 0, stream>>>(xl, xr, src, dst,
                                                     att + (size_t)l * H_DIM * C_DIM,
                                                     alpha, amax, E);
        edge_scatter<<<edgeBlocks, 256, 0, stream>>>(xl, src, dst, alpha, amax,
                                                     denom, accum, E);

        float* xout = (l == 0) ? x1 : (float*)d_out;
        finalize<<<ndBlocks, 256, 0, stream>>>(accum, denom, xres, xout, nd);
        xin = x1;
    }
}
